// QuantizedMatryoshkaDecoder_54468775247876
// MI455X (gfx1250) — matrix operand order repaired and checked by reading, not compiled, not run on hardware
//
#include <hip/hip_runtime.h>
#include <math.h>

typedef __attribute__((ext_vector_type(16))) _Float16 v16h;
typedef __attribute__((ext_vector_type(8)))  float    v8f;
typedef __attribute__((ext_vector_type(4)))  float    v4f;

static constexpr int IN_F  = 16384;
static constexpr int OUT_F = 1024;
static constexpr int BATCH = 4096;

// group id of feature k: boundaries at 128<<g
__device__ __forceinline__ int group_of(int k) {
    return (k < 128) ? 0 : (31 - __clz(k) - 6);
}

// ---------------------------------------------------------------------------
// K1: per-row scale[k] = 2^(6-g) * 0.03125 / (||Weff[k,:]||_2 + 1e-8)
// ---------------------------------------------------------------------------
__global__ __launch_bounds__(256)
void qmd_scale_kernel(const float* __restrict__ w, const float* __restrict__ wm,
                      float* __restrict__ scale) {
    __shared__ float red[256];
    const int k = blockIdx.x;
    const int t = threadIdx.x;
    float s = 0.f;
    const size_t rowb = (size_t)k * OUT_F;
    for (int n = t; n < OUT_F; n += 256) {
        float a = w[rowb + n];
        float b = wm[rowb + n];
        float eff = ((a >= 0.f) ? 1.f : -1.f) + ((b >= 0.f) ? 1.f : -1.f);
        s += eff * eff;
    }
    red[t] = s;
    __syncthreads();
    for (int o = 128; o > 0; o >>= 1) {
        if (t < o) red[t] += red[t + o];
        __syncthreads();
    }
    if (t == 0) {
        float l2 = sqrtf(red[0]);
        int g = group_of(k);
        scale[k] = exp2f((float)(6 - g)) * 0.03125f / (l2 + 1e-8f);
    }
}

// ---------------------------------------------------------------------------
// K2: BT[n][k] = (half)(scale[k] * Weff(k,n)), N-major, via LDS transpose
// ---------------------------------------------------------------------------
__global__ __launch_bounds__(256)
void qmd_bt_kernel(const float* __restrict__ w, const float* __restrict__ wm,
                   const float* __restrict__ scale, _Float16* __restrict__ BT) {
    __shared__ _Float16 tile[32][33];
    const int k0 = blockIdx.x * 32;
    const int n0 = blockIdx.y * 32;
    const int t  = threadIdx.x;

    {   // load phase: thread covers (kk = t>>3, nn = (t&7)*4 .. +3), coalesced in n
        const int kk  = t >> 3;
        const int nnb = (t & 7) * 4;
        const float sc = scale[k0 + kk];
        const size_t base = (size_t)(k0 + kk) * OUT_F + (n0 + nnb);
        #pragma unroll
        for (int e = 0; e < 4; ++e) {
            float a = w[base + e];
            float b = wm[base + e];
            float eff = ((a >= 0.f) ? 1.f : -1.f) + ((b >= 0.f) ? 1.f : -1.f);
            tile[kk][nnb + e] = (_Float16)(sc * eff);
        }
    }
    __syncthreads();
    {   // store phase: thread covers (nn = t>>3, kk = (t&7)*4 .. +3), coalesced in k
        const int nn  = t >> 3;
        const int kkb = (t & 7) * 4;
        const size_t obase = (size_t)(n0 + nn) * IN_F + (k0 + kkb);
        #pragma unroll
        for (int e = 0; e < 4; ++e)
            BT[obase + e] = tile[kkb + e][nn];
    }
}

// ---------------------------------------------------------------------------
// K3: group sums of binarized latent (each 16-elem chunk lies in one group)
// ---------------------------------------------------------------------------
__global__ __launch_bounds__(256)
void qmd_zsum_kernel(const float* __restrict__ latent, float* __restrict__ gsum) {
    __shared__ float gs[8];
    const int t = threadIdx.x;
    if (t < 8) gs[t] = 0.f;
    __syncthreads();
    const size_t chunk = (size_t)blockIdx.x * 256 + t;
    const size_t base  = chunk * 16;
    const int k = (int)(base % IN_F);
    const int g = group_of(k);
    float s = 0.f;
    #pragma unroll
    for (int e = 0; e < 16; ++e)
        s += (latent[base + e] > 0.5f) ? 1.f : 0.f;
    atomicAdd(&gs[g], s);
    __syncthreads();
    if (t < 8) atomicAdd(&gsum[t], gs[t]);
}

__global__ void qmd_zero_kernel(float* p) {
    if (threadIdx.x < 8) p[threadIdx.x] = 0.f;
}

__global__ void qmd_fin_kernel(const float* __restrict__ gsum, float* __restrict__ out) {
    if (threadIdx.x < 8) out[threadIdx.x] = gsum[threadIdx.x] * (1.f / (float)BATCH);
}

// ---------------------------------------------------------------------------
// K4: GEMM with 8 cumulative prefix snapshots.
// 8 waves/block arranged 4(M) x 2(N); wave tile = 32(M) x 128(N)
// -> block tile = 128 x 256; 16 v_wmma per K-step of 32.
// A built on-the-fly from f32 latent (threshold 0.5 -> {0,1} f16).
// B loaded from N-major BT: one contiguous aligned 32B chunk per lane/frag.
// ---------------------------------------------------------------------------
__global__ __launch_bounds__(256)
void qmd_gemm_kernel(const float* __restrict__ latent,
                     const _Float16* __restrict__ BT,
                     const float* __restrict__ bias,
                     float* __restrict__ out) {
    constexpr int MI = 2;   // 16-row A subtiles per wave
    constexpr int NI = 8;   // 16-col B subtiles per wave

    const int lane  = threadIdx.x & 31;
    const int wave  = threadIdx.x >> 5;
    const int wm    = wave >> 1;          // 0..3
    const int wn    = wave & 1;           // 0..1
    const int m0    = blockIdx.y * 128 + wm * 32;
    const int n0    = blockIdx.x * 256 + wn * 128;

    const int lrow  = lane & 15;          // N (for B/C) or M (for A)
    const int lhalf = lane >> 4;          // 0 or 1
    const int koffA = lhalf * 16;         // A: K sub-offset per half-wave
    const int koffB = lhalf * 16;         // B: K sub-offset per half-wave

    const _Float16 h1 = (_Float16)1.0f;
    const _Float16 h0 = (_Float16)0.0f;

    v8f acc[MI][NI];
    #pragma unroll
    for (int mi = 0; mi < MI; ++mi)
        #pragma unroll
        for (int ni = 0; ni < NI; ++ni)
            acc[mi][ni] = (v8f){0.f, 0.f, 0.f, 0.f, 0.f, 0.f, 0.f, 0.f};

    float biasv[NI];
    #pragma unroll
    for (int ni = 0; ni < NI; ++ni)
        biasv[ni] = bias[n0 + ni * 16 + lrow];

    // per-mi base pointers for the latent rows this half-lane reads
    const float* arow[MI];
    #pragma unroll
    for (int mi = 0; mi < MI; ++mi)
        arow[mi] = latent + (size_t)(m0 + mi * 16 + lrow) * IN_F + koffA;

    float* __restrict__ res = out + 8;    // results base after 8 group scalars

    int kPrev = 0;
    for (int g = 0; g < 8; ++g) {
        const int kEndG = 128 << g;
        for (int k0 = kPrev; k0 < kEndG; k0 += 32) {
            // prefetch next K-tile of the streaming A operand (global_prefetch_b8)
            #pragma unroll
            for (int mi = 0; mi < MI; ++mi)
                __builtin_prefetch(arow[mi] + k0 + 32, 0, 1);

            // A fragments: lane holds row m0+mi*16+lrow, K = koffA .. +15 contiguous
            v16h afrag[MI];
            #pragma unroll
            for (int mi = 0; mi < MI; ++mi) {
                const float* ap = arow[mi] + k0;
                v4f f0 = *(const v4f*)(ap);
                v4f f1 = *(const v4f*)(ap + 4);
                v4f f2 = *(const v4f*)(ap + 8);
                v4f f3 = *(const v4f*)(ap + 12);
                v16h a;
                #pragma unroll
                for (int j = 0; j < 4; ++j) {
                    a[j]      = (f0[j] > 0.5f) ? h1 : h0;
                    a[4 + j]  = (f1[j] > 0.5f) ? h1 : h0;
                    a[8 + j]  = (f2[j] > 0.5f) ? h1 : h0;
                    a[12 + j] = (f3[j] > 0.5f) ? h1 : h0;
                }
                afrag[mi] = a;
            }
            // B fragments: lane holds col n0+ni*16+lrow, K = k0+koffB .. +15 contiguous
            v16h bfrag[NI];
            #pragma unroll
            for (int ni = 0; ni < NI; ++ni) {
                const _Float16* bp = BT + (size_t)(n0 + ni * 16 + lrow) * IN_F + k0 + koffB;
                bfrag[ni] = *(const v16h*)bp;
            }
            #pragma unroll
            for (int mi = 0; mi < MI; ++mi)
                #pragma unroll
                for (int ni = 0; ni < NI; ++ni)
                    acc[mi][ni] = __builtin_amdgcn_wmma_f32_16x16x32_f16(
                        false, afrag[mi], false, bfrag[ni],
                        (short)0, acc[mi][ni], false, false);
        }
        // cumulative snapshot for group g (+bias, carried from group 0)
        float* __restrict__ resg = res + (size_t)g * BATCH * OUT_F;
        #pragma unroll
        for (int mi = 0; mi < MI; ++mi) {
            #pragma unroll
            for (int ni = 0; ni < NI; ++ni) {
                const int ncol = n0 + ni * 16 + lrow;
                #pragma unroll
                for (int r = 0; r < 8; ++r) {
                    const int mrow = m0 + mi * 16 + lhalf * 8 + r;
                    resg[(size_t)mrow * OUT_F + ncol] = acc[mi][ni][r] + biasv[ni];
                }
            }
        }
        kPrev = kEndG;
    }
}

// ---------------------------------------------------------------------------
// Launch. Workspace layout (needs ~33.7 MB):
//   [0, 32)              : 8 f32 group-sum accumulators
//   [256, 256+64K)       : scale[16384] f32
//   [65792, 65792+32M)   : BT[1024][16384] f16 (N-major)
// ---------------------------------------------------------------------------
extern "C" void kernel_launch(void* const* d_in, const int* in_sizes, int n_in,
                              void* d_out, int out_size, void* d_ws, size_t ws_size,
                              hipStream_t stream) {
    const float* latent = (const float*)d_in[0];
    const float* weight = (const float*)d_in[1];
    const float* wmir   = (const float*)d_in[2];
    const float* bias   = (const float*)d_in[3];
    float* out = (float*)d_out;

    char* ws = (char*)d_ws;
    float*    gsum  = (float*)ws;
    float*    scale = (float*)(ws + 256);
    _Float16* BT    = (_Float16*)(ws + 256 + (size_t)IN_F * sizeof(float));

    qmd_zero_kernel<<<1, 32, 0, stream>>>(gsum);
    qmd_scale_kernel<<<IN_F, 256, 0, stream>>>(weight, wmir, scale);
    qmd_bt_kernel<<<dim3(IN_F / 32, OUT_F / 32), 256, 0, stream>>>(weight, wmir, scale, BT);
    qmd_zsum_kernel<<<(BATCH * IN_F / 16) / 256, 256, 0, stream>>>(latent, gsum);
    qmd_fin_kernel<<<1, 32, 0, stream>>>(gsum, out);
    qmd_gemm_kernel<<<dim3(OUT_F / 256, BATCH / 128), 256, 0, stream>>>(latent, BT, bias, out);
}
